// Model_1065151889489
// MI455X (gfx1250) — matrix operand order repaired and checked by reading, not compiled, not run on hardware
//
#include <hip/hip_runtime.h>
#include <math.h>

#define T_SEQ   100
#define BATCH   4096
#define D_IN    64
#define H_DIM   128
#define BT      32            // batch rows per workgroup
#define NTHREAD 256           // 8 waves (wave32)

#define LDA        264        // f16 staging row stride (256 + pad), 16B-aligned rows
#define LDH        132        // f32 row stride for h (128 + pad)
#define LDS64      68         // f32 row stride for 64-wide buffers

typedef _Float16 h16;
typedef __attribute__((ext_vector_type(16))) _Float16 v16h;
typedef __attribute__((ext_vector_type(8)))  _Float16 v8h;
typedef __attribute__((ext_vector_type(8)))  float    v8f;

// ---------------- workspace float layout ----------------
#define WSF_XL    0
#define WSF_YNUM  1
#define WSF_YDEN  2
#define WSF_DENOM 4       // 100 (stores 1/(sum+1e-5))
#define WSF_BG    104     // 512  (b_ih + b_hh)
#define WSF_BDH   616     // 128
#define WSF_BDX   744     // 64
#define WSF_BHIST 808     // 64
#define WSF_BFEAT 872     // 64
#define WSF_BCOMB 936     // 64
#define WSF_WDXD  1000    // 64  diag(W_dx)
#define WSF_WOUT  1064    // 128
#define WSF_BOUT  1192    // 1
#define WSF_COUNT 1216
#define FRAG_BYTE_OFF (WSF_COUNT * 4)   // 4864 bytes, 64-aligned

// fragment regions (each fragment = 512 halfs = 32 lanes x 16 halfs)
#define FR_WDH    0       // 16  (K=64 , N=128)  d @ W_dh^T
#define FR_WHIST  16      // 16  (K=128, N=64 )  h @ W_hist^T
#define FR_WFEAT  32      // 8   (K=64 , N=64 )  x_c @ Wfeat^T (zero diag)
#define FR_WCOMB  40      // 16  (K=128, N=64 )  [gx,m] @ W_comb^T
#define FR_WG     56      // 256 (K=256, N=512)  [c_c,m,h] @ [W_ih;W_hh]^T
#define FR_TOTAL  312

// dynamic LDS: sH + 5 small f32 bufs + f16 staging + biases + reduce  (~82KB -> 2-3 WGs/WGP)
#define SMEM_BYTES ((BT*LDH + 5*BT*LDS64)*4 + BT*LDA*2 + 960*4 + 256*4)  // 82176

// ---------------- fast transcendentals (native v_exp_f32 / v_rcp_f32) ----------------
#if __has_builtin(__builtin_amdgcn_rcpf)
#define FRCP(x) __builtin_amdgcn_rcpf(x)
#else
#define FRCP(x) (1.f / (x))
#endif
__device__ __forceinline__ float fexp(float x) { return __expf(x); }
__device__ __forceinline__ float fsig(float x) { return FRCP(1.f + __expf(-x)); }
__device__ __forceinline__ float ftanh(float x) {
  x = fminf(fmaxf(x, -15.f), 15.f);
  float e = __expf(2.f * x);
  return (e - 1.f) * FRCP(e + 1.f);
}

// ---------------- fragment helpers (contiguous k order: lane half h holds k 16h..16h+15) ----------------
__device__ __forceinline__ v16h load_a_frag(const h16* sA, int mt, int kbase, int lane) {
  // 16-bit A 16x32: lanes 0-15 row M=lane hold K 0..15; lanes 16-31 row M=lane-16 hold K 16..31
  int m = (mt << 4) + (lane & 15);
  const h16* p = sA + m * LDA + kbase + ((lane & 16) ? 16 : 0);
  v8h lo = *(const v8h*)p;
  v8h hi = *(const v8h*)(p + 8);
  return __builtin_shufflevector(lo, hi, 0,1,2,3,4,5,6,7,8,9,10,11,12,13,14,15);
}
__device__ __forceinline__ v16h load_b_frag(const h16* fragBase, int idx, int lane) {
  // pre-swizzled: 32 contiguous bytes per lane, 1KB per wave-fragment
  const h16* p = fragBase + ((size_t)idx << 9) + (lane << 4);
  v8h lo = *(const v8h*)p;
  v8h hi = *(const v8h*)(p + 8);
  return __builtin_shufflevector(lo, hi, 0,1,2,3,4,5,6,7,8,9,10,11,12,13,14,15);
}
__device__ __forceinline__ v8f wmma16(v16h a, v16h b, v8f c) {
  return __builtin_amdgcn_wmma_f32_16x16x32_f16(false, a, false, b, (short)0, c, false, false);
}

// ---------------- prep: weights -> f16 swizzled fragments + biases ----------------
__global__ void prep_kernel(const float* __restrict__ W_ih, const float* __restrict__ W_hh,
                            const float* __restrict__ b_ih, const float* __restrict__ b_hh,
                            const float* __restrict__ W_dh, const float* __restrict__ b_dh,
                            const float* __restrict__ W_dx, const float* __restrict__ b_dx,
                            const float* __restrict__ W_hist, const float* __restrict__ b_hist,
                            const float* __restrict__ W_feat, const float* __restrict__ b_feat,
                            const float* __restrict__ W_comb, const float* __restrict__ b_comb,
                            const float* __restrict__ W_out, const float* __restrict__ b_out,
                            float* wsf, h16* frags) {
  int f = blockIdx.x;
  int l = threadIdx.x;                 // 0..31
  if (f < FR_TOTAL) {
    int n16   = l & 15;
    int khalf = (l & 16) ? 16 : 0;     // B 32x16: lanes 0-15 -> K 0..15, lanes 16-31 -> K 16..31
    h16* out = frags + (size_t)f * 512 + l * 16;
    if (f < FR_WHIST) {                               // W_dh^T : B[k][n] = W_dh[n*64+k]
      int lo = f, kt = lo >> 3, nt = lo & 7, n = nt * 16 + n16;
      for (int i = 0; i < 16; ++i) { int k = kt * 32 + khalf + i; out[i] = (h16)W_dh[n * 64 + k]; }
    } else if (f < FR_WFEAT) {                        // W_hist^T : B[k][n] = W_hist[n*128+k]
      int lo = f - FR_WHIST, kt = lo >> 2, nt = lo & 3, n = nt * 16 + n16;
      for (int i = 0; i < 16; ++i) { int k = kt * 32 + khalf + i; out[i] = (h16)W_hist[n * 128 + k]; }
    } else if (f < FR_WCOMB) {                        // Wfeat^T zero-diag
      int lo = f - FR_WFEAT, kt = lo >> 2, nt = lo & 3, n = nt * 16 + n16;
      for (int i = 0; i < 16; ++i) { int k = kt * 32 + khalf + i;
        out[i] = (h16)((k == n) ? 0.f : W_feat[n * 64 + k]); }
    } else if (f < FR_WG) {                           // W_comb^T
      int lo = f - FR_WCOMB, kt = lo >> 2, nt = lo & 3, n = nt * 16 + n16;
      for (int i = 0; i < 16; ++i) { int k = kt * 32 + khalf + i; out[i] = (h16)W_comb[n * 128 + k]; }
    } else {                                          // gates: rows 0..127 = W_ih^T, 128..255 = W_hh^T
      int lo = f - FR_WG, kt = lo >> 5, nt = lo & 31, n = nt * 16 + n16;
      for (int i = 0; i < 16; ++i) { int k = kt * 32 + khalf + i;
        out[i] = (h16)((k < 128) ? W_ih[n * 128 + k] : W_hh[n * 128 + (k - 128)]); }
    }
  } else {
    for (int j = l; j < 512; j += 32) wsf[WSF_BG + j]    = b_ih[j] + b_hh[j];
    for (int j = l; j < 128; j += 32) wsf[WSF_BDH + j]   = b_dh[j];
    for (int j = l; j < 128; j += 32) wsf[WSF_WOUT + j]  = W_out[j];
    for (int j = l; j < 64;  j += 32) {
      wsf[WSF_BDX + j]   = b_dx[j];
      wsf[WSF_BHIST + j] = b_hist[j];
      wsf[WSF_BFEAT + j] = b_feat[j];
      wsf[WSF_BCOMB + j] = b_comb[j];
      wsf[WSF_WDXD + j]  = W_dx[j * 64 + j];
    }
    if (l == 0) { wsf[WSF_BOUT] = b_out[0]; wsf[WSF_XL] = 0.f; wsf[WSF_YNUM] = 0.f; wsf[WSF_YDEN] = 0.f; }
  }
}

// ---------------- denom[t]: store reciprocal 1/(sum(masks[:,t,:])+1e-5) ----------------
__global__ void denom_kernel(const int* __restrict__ masks, float* wsf) {
  __shared__ float red[256];
  int t = blockIdx.x, tid = threadIdx.x;
  float s = 0.f;
  for (int i = tid; i < BATCH * D_IN; i += 256) {
    int b = i >> 6, d = i & 63;
    s += (float)masks[((size_t)b * T_SEQ + t) * D_IN + d];
  }
  red[tid] = s; __syncthreads();
  for (int off = 128; off > 0; off >>= 1) { if (tid < off) red[tid] += red[tid + off]; __syncthreads(); }
  if (tid == 0) wsf[WSF_DENOM + t] = 1.f / (red[0] + 1e-5f);
}

// ---------------- main recurrent kernel ----------------
// NOTE: `frags` deliberately NOT __restrict__: the per-step global stores (imps) may
// alias it as far as the compiler knows, which blocks LICM from hoisting the
// loop-invariant B-fragment loads out of the time loop (that hoisting caused
// catastrophic scratch spills). Fragments are meant to be re-read from L2 each step.
__launch_bounds__(NTHREAD)
__global__ void rits_main(const float* __restrict__ values, const int* __restrict__ masks,
                          const float* __restrict__ deltas,
                          const int* __restrict__ labels, const int* __restrict__ is_train,
                          float* wsf, const h16* frags, float* d_out) {
  extern __shared__ char smem[];
  float* sH  = (float*)smem;                 // [32][132] hidden state (f32)
  float* sX  = sH  + BT * LDH;               // [32][68] x
  float* sM  = sX  + BT * LDS64;             // m
  float* sXh = sM  + BT * LDS64;             // x_h
  float* sGx = sXh + BT * LDS64;             // gamma_x
  float* sZh = sGx + BT * LDS64;             // z_h
  h16*   sA  = (h16*)(sZh + BT * LDS64);     // [32][264] f16 A staging
  float* sB  = (float*)(sA + BT * LDA);      // biases: [0:512]=bg [512:640]=bdh [640:704]=bhist
                                             // [704:768]=bfeat [768:832]=bcomb [832:896]=wdxd [896:960]=bdx
  float* sRed = sB + 960;                    // [256]

  const int tid  = threadIdx.x;
  const int wave = tid >> 5, lane = tid & 31;
  const int r  = tid >> 3;                   // row 0..31 for elementwise phases
  const int cq = (tid & 7) << 3;             // 8-col chunk
  const int b0 = blockIdx.x * BT;
  const int ln15 = lane & 15;
  const int rhalf = (lane & 16) ? 8 : 0;
  float* imps = d_out + 1 + BATCH;

  for (int j = tid; j < 512; j += NTHREAD) sB[j] = wsf[WSF_BG + j];
  for (int j = tid; j < 128; j += NTHREAD) sB[512 + j] = wsf[WSF_BDH + j];
  for (int j = tid; j < 64;  j += NTHREAD) {
    sB[640 + j] = wsf[WSF_BHIST + j]; sB[704 + j] = wsf[WSF_BFEAT + j];
    sB[768 + j] = wsf[WSF_BCOMB + j]; sB[832 + j] = wsf[WSF_WDXD + j];
    sB[896 + j] = wsf[WSF_BDX + j];
  }
  for (int i = tid; i < BT * LDH; i += NTHREAD) sH[i] = 0.f;
  __syncthreads();

  // ---- hoist loop-invariant per-lane bias values into registers ----
  float wdxd_r[8], bdx_r[8];
#pragma unroll
  for (int i = 0; i < 8; ++i) { wdxd_r[i] = sB[832 + cq + i]; bdx_r[i] = sB[896 + cq + i]; }
  float bdh_r[2];
#pragma unroll
  for (int j = 0; j < 2; ++j) { int nt = (wave * 2 + j) & 7; bdh_r[j] = sB[512 + nt * 16 + ln15]; }
  const int ntw = wave & 3;                  // phases 2/3b/4: one tile per wave
  const float bhist_r = sB[640 + ntw * 16 + ln15];
  const float bfeat_r = sB[704 + ntw * 16 + ln15];
  const float bcomb_r = sB[768 + ntw * 16 + ln15];
  float bg_r[2][4];
#pragma unroll
  for (int j = 0; j < 2; ++j) {
    int ht = (wave * 2 + j) & 7, colh = ht * 16 + ln15;
#pragma unroll
    for (int gq = 0; gq < 4; ++gq) bg_r[j][gq] = sB[gq * 128 + colh];
  }
  // LSTM cell state lives entirely in registers (fixed (wave,j,g,lane) ownership)
  float creg[2][8];
#pragma unroll
  for (int j = 0; j < 2; ++j)
#pragma unroll
    for (int g = 0; g < 8; ++g) creg[j][g] = 0.f;

  float xl_acc = 0.f;

  for (int t = 0; t < T_SEQ; ++t) {
    const float invDen = wsf[WSF_DENOM + t];   // precomputed reciprocal

    // ---- phase 0: load x/m/d, gamma_x elementwise, stage d(f16) + m(f16)
    {
      const size_t base = ((size_t)(b0 + r) * T_SEQ + t) * D_IN + cq;
      const float* xp = values + base;  const int* mp = masks + base;  const float* dp = deltas + base;
#pragma unroll
      for (int i = 0; i < 8; ++i) {
        float xv = xp[i], dv = dp[i], mv = (float)mp[i];
        int c = cq + i;
        sX[r * LDS64 + c] = xv; sM[r * LDS64 + c] = mv;
        sA[r * LDA + c] = (h16)dv;               // cols 0..63  : d
        sA[r * LDA + 64 + c] = (h16)mv;          // cols 64..127: m (stable this step)
        sGx[r * LDS64 + c] = fexp(-fmaxf(dv * wdxd_r[i] + bdx_r[i], 0.f));
      }
      if (t + 1 < T_SEQ) {
        __builtin_prefetch(xp + D_IN, 0, 0);
        __builtin_prefetch(mp + D_IN, 0, 0);
        __builtin_prefetch(dp + D_IN, 0, 0);
      }
    }
    __syncthreads();

    // ---- phase 1: gamma_h = exp(-relu(d@Wdh^T+b)); h *= gamma_h; stage h(f16) cols 128..255
    for (int j = 0; j < 2; ++j) {
      int tile = wave * 2 + j, mt = tile >> 3, nt = tile & 7;
      v8f acc = {};
      for (int kt = 0; kt < 2; ++kt)
        acc = wmma16(load_a_frag(sA, mt, kt * 32, lane),
                     load_b_frag(frags, FR_WDH + kt * 8 + nt, lane), acc);
      int col = nt * 16 + ln15;
      int row0 = mt * 16 + rhalf;
#pragma unroll
      for (int g = 0; g < 8; ++g) {
        int row = row0 + g;
        float gh = fexp(-fmaxf(acc[g] + bdh_r[j], 0.f));
        float hv = sH[row * LDH + col] * gh;
        sH[row * LDH + col] = hv;
        sA[row * LDA + 128 + col] = (h16)hv;
      }
    }
    __syncthreads();

    // ---- phase 2: x_h = h @ W_hist^T + b_hist (one 16x16 tile per wave)
    {
      int mt = wave >> 2, nt = ntw;
      v8f acc = {};
      for (int kt = 0; kt < 4; ++kt)
        acc = wmma16(load_a_frag(sA, mt, 128 + kt * 32, lane),
                     load_b_frag(frags, FR_WHIST + kt * 4 + nt, lane), acc);
      int col = nt * 16 + ln15;
      int row0 = mt * 16 + rhalf;
#pragma unroll
      for (int g = 0; g < 8; ++g) sXh[(row0 + g) * LDS64 + col] = acc[g] + bhist_r;
    }
    __syncthreads();

    // ---- phase 3a: x_c = m*x+(1-m)*x_h -> f16 cols 0..63; loss term 1
#pragma unroll
    for (int i = 0; i < 8; ++i) {
      int c = cq + i;
      float xv = sX[r * LDS64 + c], mv = sM[r * LDS64 + c], xh = sXh[r * LDS64 + c];
      sA[r * LDA + c] = (h16)(mv * xv + (1.f - mv) * xh);
      xl_acc += fabsf(xv - xh) * mv * invDen;
    }
    __syncthreads();

    // ---- phase 3b: z_h = x_c @ Wfeat^T + b_feat (one tile per wave)
    {
      int mt = wave >> 2, nt = ntw;
      v8f acc = {};
      for (int kt = 0; kt < 2; ++kt)
        acc = wmma16(load_a_frag(sA, mt, kt * 32, lane),
                     load_b_frag(frags, FR_WFEAT + kt * 4 + nt, lane), acc);
      int col = nt * 16 + ln15;
      int row0 = mt * 16 + rhalf;
#pragma unroll
      for (int g = 0; g < 8; ++g) sZh[(row0 + g) * LDS64 + col] = acc[g] + bfeat_r;
    }
    __syncthreads();

    // ---- phase 3c: stage gamma_x (f16) into cols 0..63
#pragma unroll
    for (int i = 0; i < 8; ++i) { int c = cq + i; sA[r * LDA + c] = (h16)sGx[r * LDS64 + c]; }
    __syncthreads();

    // ---- phase 4: alpha = [gx,m] @ Wcomb^T + b; c_h, c_c, losses 2&3, imputations, stage c_c(f16)
    {
      int mt = wave >> 2, nt = ntw;
      v8f acc = {};
      for (int kt = 0; kt < 4; ++kt)
        acc = wmma16(load_a_frag(sA, mt, kt * 32, lane),
                     load_b_frag(frags, FR_WCOMB + kt * 4 + nt, lane), acc);
      int col = nt * 16 + ln15;
      int row0 = mt * 16 + rhalf;
#pragma unroll
      for (int g = 0; g < 8; ++g) {
        int row = row0 + g;
        float al = acc[g] + bcomb_r;
        float zh = sZh[row * LDS64 + col], xh = sXh[row * LDS64 + col];
        float ch = al * zh + (1.f - al) * xh;
        float xv = sX[row * LDS64 + col], mv = sM[row * LDS64 + col];
        xl_acc += (fabsf(xv - zh) + fabsf(xv - ch)) * mv * invDen;
        float cc = mv * xv + (1.f - mv) * ch;
        sA[row * LDA + col] = (h16)cc;
        imps[((size_t)(b0 + row) * T_SEQ + t) * D_IN + col] = cc;
      }
    }
    __syncthreads();

    // ---- phase 5: gates = [c_c,m,h] @ [W_ih;W_hh]^T + bg; LSTM update, c in registers
    for (int j = 0; j < 2; ++j) {
      int combo = wave * 2 + j, mt = combo >> 3, ht = combo & 7;
      v8f ai = {}, af = {}, ag = {}, ao = {};
      for (int kt = 0; kt < 8; ++kt) {
        v16h a = load_a_frag(sA, mt, kt * 32, lane);
        ai = wmma16(a, load_b_frag(frags, FR_WG + kt * 32 + 0 * 8 + ht, lane), ai);
        af = wmma16(a, load_b_frag(frags, FR_WG + kt * 32 + 1 * 8 + ht, lane), af);
        ag = wmma16(a, load_b_frag(frags, FR_WG + kt * 32 + 2 * 8 + ht, lane), ag);
        ao = wmma16(a, load_b_frag(frags, FR_WG + kt * 32 + 3 * 8 + ht, lane), ao);
      }
      int colh = ht * 16 + ln15;
      int row0 = mt * 16 + rhalf;
#pragma unroll
      for (int g = 0; g < 8; ++g) {
        int row = row0 + g;
        float iv = fsig(ai[g] + bg_r[j][0]);
        float fv = fsig(af[g] + bg_r[j][1]);
        float gv = ftanh(ag[g] + bg_r[j][2]);
        float ov = fsig(ao[g] + bg_r[j][3]);
        float cn = fv * creg[j][g] + iv * gv;
        creg[j][g] = cn;
        sH[row * LDH + colh] = ov * ftanh(cn);
      }
    }
    __syncthreads();
  }

  // ---- output head: y = h @ W_out^T + b_out; BCE; reductions
  {
    int q = tid & 7;                    // 8 partial-dot threads per row
    float part = 0.f;
    for (int j = q * 16; j < q * 16 + 16; ++j) part += sH[r * LDH + j] * wsf[WSF_WOUT + j];
    sRed[tid] = part; __syncthreads();
    float ynum_t = 0.f, yden_t = 0.f;
    if (q == 0) {
      float y = wsf[WSF_BOUT];
#pragma unroll
      for (int k = 0; k < 8; ++k) y += sRed[tid + k];
      d_out[1 + b0 + r] = fsig(y);
      float lab = (float)labels[b0 + r];
      float tr  = (float)is_train[b0 + r];
      float bce = fmaxf(y, 0.f) - y * lab + __logf(1.f + __expf(-fabsf(y)));
      ynum_t = bce * tr; yden_t = tr;
    }
    __syncthreads();
    sRed[tid] = xl_acc; __syncthreads();
    for (int off = 128; off > 0; off >>= 1) { if (tid < off) sRed[tid] += sRed[tid + off]; __syncthreads(); }
    if (tid == 0) atomicAdd(&wsf[WSF_XL], sRed[0]);
    __syncthreads();
    sRed[tid] = ynum_t; __syncthreads();
    for (int off = 128; off > 0; off >>= 1) { if (tid < off) sRed[tid] += sRed[tid + off]; __syncthreads(); }
    if (tid == 0) atomicAdd(&wsf[WSF_YNUM], sRed[0]);
    __syncthreads();
    sRed[tid] = yden_t; __syncthreads();
    for (int off = 128; off > 0; off >>= 1) { if (tid < off) sRed[tid] += sRed[tid + off]; __syncthreads(); }
    if (tid == 0) atomicAdd(&wsf[WSF_YDEN], sRed[0]);
  }
}

__global__ void final_kernel(const float* __restrict__ wsf, float* __restrict__ d_out) {
  if (threadIdx.x == 0 && blockIdx.x == 0)
    d_out[0] = wsf[WSF_XL] * 0.3f + (wsf[WSF_YNUM] / (wsf[WSF_YDEN] + 1e-5f)) * 1.0f;
}

extern "C" void kernel_launch(void* const* d_in, const int* in_sizes, int n_in,
                              void* d_out, int out_size, void* d_ws, size_t ws_size,
                              hipStream_t stream) {
  (void)in_sizes; (void)n_in; (void)out_size; (void)ws_size;
  const float* values  = (const float*)d_in[0];
  const int*   masks   = (const int*)d_in[1];
  const float* deltas  = (const float*)d_in[2];
  const int*   labels  = (const int*)d_in[5];
  const int*   is_trn  = (const int*)d_in[6];
  const float* W_ih  = (const float*)d_in[7];
  const float* W_hh  = (const float*)d_in[8];
  const float* b_ih  = (const float*)d_in[9];
  const float* b_hh  = (const float*)d_in[10];
  const float* W_dh  = (const float*)d_in[11];
  const float* b_dh  = (const float*)d_in[12];
  const float* W_dx  = (const float*)d_in[13];
  const float* b_dx  = (const float*)d_in[14];
  const float* W_hist = (const float*)d_in[15];
  const float* b_hist = (const float*)d_in[16];
  const float* W_feat = (const float*)d_in[17];
  const float* b_feat = (const float*)d_in[18];
  const float* W_comb = (const float*)d_in[19];
  const float* b_comb = (const float*)d_in[20];
  const float* W_out  = (const float*)d_in[21];
  const float* b_out  = (const float*)d_in[22];

  float* wsf  = (float*)d_ws;
  h16*  frags = (h16*)((char*)d_ws + FRAG_BYTE_OFF);
  float* out  = (float*)d_out;

  prep_kernel<<<FR_TOTAL + 1, 32, 0, stream>>>(W_ih, W_hh, b_ih, b_hh, W_dh, b_dh, W_dx, b_dx,
                                               W_hist, b_hist, W_feat, b_feat, W_comb, b_comb,
                                               W_out, b_out, wsf, frags);
  denom_kernel<<<T_SEQ, 256, 0, stream>>>(masks, wsf);

  (void)hipFuncSetAttribute((const void*)rits_main,
                            hipFuncAttributeMaxDynamicSharedMemorySize, SMEM_BYTES);
  rits_main<<<BATCH / BT, NTHREAD, SMEM_BYTES, stream>>>(values, masks, deltas, labels, is_trn,
                                                         wsf, frags, out);
  final_kernel<<<1, 32, 0, stream>>>(wsf, out);
}
